// GCN_52913997086747
// MI455X (gfx1250) — matrix operand order repaired and checked by reading, not compiled, not run on hardware
//
#include <hip/hip_runtime.h>

typedef __attribute__((ext_vector_type(16))) _Float16 v16h;
typedef __attribute__((ext_vector_type(8)))  _Float16 v8h;
typedef __attribute__((ext_vector_type(4)))  _Float16 v4h;
typedef __attribute__((ext_vector_type(2)))  _Float16 v2h;
typedef __attribute__((ext_vector_type(8)))  float    v8f;
typedef __attribute__((ext_vector_type(4)))  float    v4f;

static constexpr int EDIM = 128;   // embedding dim
static constexpr int RELS = 8;     // relations

// ---------------------------------------------------------------------------
// Pack W[r][i][h] (fp32) into Wp[r][kk][i] = {f16(W[r][i][2kk]), f16(W[r][i][2kk+1])}
// This is exactly the B-fragment dword layout for v_wmma_f32_16x16x32_f16.
// ---------------------------------------------------------------------------
__global__ void rgcn_pack_w(const float* __restrict__ W,
                            unsigned int* __restrict__ Wp, int total) {
    int t = blockIdx.x * blockDim.x + threadIdx.x;
    if (t >= total) return;
    int i  = t & (EDIM - 1);        // output col, fastest
    int kk = (t >> 7) & 63;         // K pair index
    int r  = t >> 13;               // relation
    const float* wr = W + (size_t)r * EDIM * EDIM + (size_t)i * EDIM + 2 * kk;
    union { v2h h2; unsigned int u; } u;
    u.h2.x = (_Float16)wr[0];       // even K -> [15:0]
    u.h2.y = (_Float16)wr[1];       // odd  K -> [31:16]
    Wp[t] = u.u;
}

// ---------------------------------------------------------------------------
// deg[rel*N + dst] += 1   (fp32 atomics; deg zeroed beforehand)
// ---------------------------------------------------------------------------
__global__ void rgcn_degree(const int* __restrict__ erel,
                            const int* __restrict__ edst,
                            float* __restrict__ deg, int E, int N) {
    int t = blockIdx.x * blockDim.x + threadIdx.x;
    if (t >= E) return;
    int row = erel[t] * N + edst[t];
    (void)__hip_atomic_fetch_add(&deg[row], 1.0f,
                                 __ATOMIC_RELAXED, __HIP_MEMORY_SCOPE_AGENT);
}

// ---------------------------------------------------------------------------
// One wave (32 lanes) per edge: h[row] += (1/deg[row]) * x[src]  (128 floats)
// Lane l handles floats [4l, 4l+4): float4 gather + 4 f32 global atomics.
// ---------------------------------------------------------------------------
__global__ void rgcn_scatter(const float* __restrict__ x,
                             const int* __restrict__ erel,
                             const int* __restrict__ esrc,
                             const int* __restrict__ edst,
                             const float* __restrict__ deg,
                             float* __restrict__ h, int E, int N) {
    int gid  = blockIdx.x * blockDim.x + threadIdx.x;
    int wid  = gid >> 5;
    int lane = gid & 31;
    if (wid >= E) return;
    int rr = erel[wid];
    int ss = esrc[wid];
    int dd = edst[wid];
    int row = rr * N + dd;
    float val = 1.0f / deg[row];
    const float4* xs = (const float4*)(x + (size_t)ss * EDIM);
    float4 v = xs[lane];
    float* hp = h + (size_t)row * EDIM + lane * 4;
    (void)__hip_atomic_fetch_add(hp + 0, val * v.x, __ATOMIC_RELAXED, __HIP_MEMORY_SCOPE_AGENT);
    (void)__hip_atomic_fetch_add(hp + 1, val * v.y, __ATOMIC_RELAXED, __HIP_MEMORY_SCOPE_AGENT);
    (void)__hip_atomic_fetch_add(hp + 2, val * v.z, __ATOMIC_RELAXED, __HIP_MEMORY_SCOPE_AGENT);
    (void)__hip_atomic_fetch_add(hp + 3, val * v.w, __ATOMIC_RELAXED, __HIP_MEMORY_SCOPE_AGENT);
}

// ---------------------------------------------------------------------------
// out[n,i] = relu( sum_r sum_k W[r][i][k] * h[r][n][k] )
// Block: 256 threads (8 waves). Block b handles node rows [16b, 16b+16);
// wave w handles output cols [16w, 16w+16). Per relation we stage the fp32
// 16x128 h-chunk into LDS as f16 (row stride 136 halves -> conflict-free
// b128 A reads) and the 64x128 packed W pairs (stride 132 dwords, 16B
// aligned rows -> b128 staging, conflict-free B reads), then issue
// 4x v_wmma_f32_16x16x32_f16 per relation with a persistent f32 accumulator.
// h is streamed once -> non-temporal loads; out is write-only -> NT stores.
// ---------------------------------------------------------------------------
__global__ __launch_bounds__(256) void rgcn_gemm(const float* __restrict__ h,
                                                 const unsigned int* __restrict__ Wp,
                                                 float* __restrict__ out, int N) {
    __shared__ alignas(16) _Float16     sA[16 * 136];   // 4.25 KB
    __shared__ alignas(16) unsigned int sB[64 * 132];   // 33.8 KB

    const int t    = threadIdx.x;
    const int wave = t >> 5;
    const int lane = t & 31;
    const int sel  = lane >> 4;     // lane half (0/1)
    const int nc   = lane & 15;     // matrix row (A) / col (B,C) within tile
    const int n0   = blockIdx.x * 16;
    const int i0   = wave * 16;

    // staging indices for the h chunk (always-safe clamped addresses; only
    // output stores are guarded, so the global loads stay unpredicated)
    const int srow = t >> 5;        // 0..7 (second pass adds 8)
    const int sc4  = t & 31;        // float4 column 0..31
    const int rowA = (n0 + srow     < N) ? srow     : 0;
    const int rowB = (n0 + srow + 8 < N) ? srow + 8 : 0;

    v8f acc = {};

    for (int r = 0; r < RELS; ++r) {
        __syncthreads();
        // ---- stage h chunk: 16 rows x 128 fp32 -> f16 in LDS (NT stream) ----
        const v4f* hsrc = (const v4f*)(h + ((size_t)r * N + n0) * EDIM);
        v4f v0 = __builtin_nontemporal_load(&hsrc[rowA * 32 + sc4]);
        v4f v1 = __builtin_nontemporal_load(&hsrc[rowB * 32 + sc4]);
        v4h a0, a1;
        a0.x = (_Float16)v0.x; a0.y = (_Float16)v0.y;
        a0.z = (_Float16)v0.z; a0.w = (_Float16)v0.w;
        a1.x = (_Float16)v1.x; a1.y = (_Float16)v1.y;
        a1.z = (_Float16)v1.z; a1.w = (_Float16)v1.w;
        *(v4h*)(&sA[(srow)     * 136 + sc4 * 4]) = a0;
        *(v4h*)(&sA[(srow + 8) * 136 + sc4 * 4]) = a1;

        // ---- stage Wp[r]: 64 x 128 dwords as uint4 (L2-resident, temporal) ----
        const uint4* wsrc = (const uint4*)(Wp + (size_t)r * (64 * 128));
#pragma unroll
        for (int q8 = 0; q8 < 8; ++q8) {
            int q  = q8 * 256 + t;      // uint4 index 0..2047
            int kk = q >> 5;            // K-pair row
            int i4 = q & 31;            // uint4 column
            uint4 wv = wsrc[q];
            *(uint4*)(&sB[kk * 132 + i4 * 4]) = wv;
        }
        __syncthreads();

        // ---- prefetch next relation's h chunk into near caches (WGP scope) ----
        if (r + 1 < RELS) {
            const v4f* hnext = (const v4f*)(h + ((size_t)(r + 1) * N + n0) * EDIM);
            __builtin_prefetch(&hnext[rowA * 32 + sc4], 0, 3);
            __builtin_prefetch(&hnext[rowB * 32 + sc4], 0, 3);
        }

#pragma unroll
        for (int k0 = 0; k0 < EDIM; k0 += 32) {
            // A fragment: row m = nc; chunks K=[k0+8*sel, +8) and K=[k0+16+8*sel, +8)
            union { v16h v; v8h h8[2]; } ua;
            const _Float16* ap = &sA[nc * 136 + k0 + 8 * sel];
            ua.h8[0] = *(const v8h*)ap;
            ua.h8[1] = *(const v8h*)(ap + 16);
            // B fragment: col i0+nc; VGPR j<4 = K pair {k0 + 8*sel + 2j, +1}, j>=4 = {k0 + 16 + 8*sel + 2(j-4), +1}
            union { v16h v; unsigned int u[8]; } ub;
            const int kkb = (k0 >> 1) + 4 * sel;
#pragma unroll
            for (int j = 0; j < 8; ++j)
                ub.u[j] = sB[(kkb + j + (j >= 4 ? 4 : 0)) * 132 + (i0 + nc)];

            acc = __builtin_amdgcn_wmma_f32_16x16x32_f16(
                false, ua.v, false, ub.v, (short)0, acc, false, false);
        }
    }

    // ---- ReLU + NT store: C VGPR j -> row n0 + j + 8*sel, col i0 + nc ----
    float* op = out + (size_t)n0 * EDIM + i0 + nc;
#pragma unroll
    for (int j = 0; j < 8; ++j) {
        int rown = j + 8 * sel;
        if (n0 + rown < N) {
            float v = acc[j];
            v = v > 0.0f ? v : 0.0f;
            __builtin_nontemporal_store(v, &op[(size_t)rown * EDIM]);
        }
    }
}

// ---------------------------------------------------------------------------
// Launcher. ws layout: [deg: R*N f32][Wp: R*64*128 u32][h: R*N*128 f32]
// ---------------------------------------------------------------------------
extern "C" void kernel_launch(void* const* d_in, const int* in_sizes, int n_in,
                              void* d_out, int out_size, void* d_ws, size_t ws_size,
                              hipStream_t stream) {
    const float* x    = (const float*)d_in[0];
    const float* W    = (const float*)d_in[1];
    const int*   erel = (const int*)d_in[2];
    const int*   esrc = (const int*)d_in[3];
    const int*   edst = (const int*)d_in[4];
    float*       out  = (float*)d_out;

    const int N = in_sizes[0] / EDIM;       // 100000
    const int E = in_sizes[2];              // 1600000

    char* ws = (char*)d_ws;
    float* deg = (float*)ws;
    size_t degBytes = (size_t)RELS * N * sizeof(float);
    size_t off = (degBytes + 255) & ~(size_t)255;
    unsigned int* Wp = (unsigned int*)(ws + off);
    size_t wpBytes = (size_t)RELS * 64 * 128 * sizeof(unsigned int);
    off += (wpBytes + 255) & ~(size_t)255;
    float* h = (float*)(ws + off);
    size_t hBytes = (size_t)RELS * N * EDIM * sizeof(float);

    hipMemsetAsync(deg, 0, degBytes, stream);
    hipMemsetAsync(h, 0, hBytes, stream);

    int packTotal = RELS * 64 * 128;
    rgcn_pack_w<<<(packTotal + 255) / 256, 256, 0, stream>>>(W, Wp, packTotal);
    rgcn_degree<<<(E + 255) / 256, 256, 0, stream>>>(erel, edst, deg, E, N);

    long scatterThreads = (long)E * 32;
    int scatterBlocks = (int)((scatterThreads + 255) / 256);
    rgcn_scatter<<<scatterBlocks, 256, 0, stream>>>(x, erel, esrc, edst, deg, h, E, N);

    int gemmBlocks = (N + 15) / 16;
    rgcn_gemm<<<gemmBlocks, 256, 0, stream>>>(h, Wp, out, N);
}
